// GAT3re_32186484916264
// MI455X (gfx1250) — matrix operand order repaired and checked by reading, not compiled, not run on hardware
//
#include <hip/hip_runtime.h>
#include <math.h>

#define NNODES 50000
#define NEDGES 800000
#define IN_C   256
#define HC12   448   // 7*64
#define OUT_C  32
#define NEG_SLOPE 0.2f

typedef __bf16 bf16_t;
typedef __attribute__((ext_vector_type(16))) __bf16 v16bf;
typedef __attribute__((ext_vector_type(8)))  __bf16 v8bf;
typedef __attribute__((ext_vector_type(8)))  float  v8f;

union ABu { v16bf v; v8bf h[2]; };

// ---------------- utility kernels ----------------

__global__ void k_fill(float* p, float v, size_t n) {
  size_t i = (size_t)blockIdx.x * blockDim.x + threadIdx.x;
  if (i < n) p[i] = v;
}

__global__ void k_f32_to_bf16(const float* in, bf16_t* out, size_t n) {
  size_t i = (size_t)blockIdx.x * blockDim.x + threadIdx.x;
  if (i < n) out[i] = (bf16_t)in[i];
}

// Pack W [K x Ncols] (row-major f32) into WMMA B-operand layout:
// tile (kt, ct): lane l holds column (l&15), K = (l>>4)*8 + i for i<8, 16 + (l>>4)*8 + (i-8) for i>=8.
__global__ void k_pack_w(const float* W, bf16_t* wpk, int K, int Ncols) {
  int kT = K >> 5, nT = Ncols >> 4;
  int idx = blockIdx.x * blockDim.x + threadIdx.x;
  int tot = kT * nT * 32;
  if (idx >= tot) return;
  int lane = idx & 31;
  int t = idx >> 5;            // t = kt*nT + ct
  int ct = t % nT;
  int kt = t / nT;
  int c  = lane & 15;
  int hs = lane >> 4;
  bf16_t* dst = wpk + (size_t)idx * 16;
#pragma unroll
  for (int i = 0; i < 16; ++i) {
    int k = (kt << 5) + (hs << 3) + (i < 8 ? i : 8 + i);
    dst[i] = (bf16_t)W[(size_t)k * Ncols + (ct << 4) + c];
  }
}

// ---------------- WMMA GEMM: D[M x Ncols] = A[M x K] * W ----------------
// One wave computes a 16 x (16*CTW) output strip: the A fragment is loaded
// once per k-tile and reused across CTW consecutive B tiles (CTW accumulators).
template <int CTW>
__global__ void __launch_bounds__(256)
k_gemm_wmma(const bf16_t* __restrict__ A, const bf16_t* __restrict__ Wpk,
            float* __restrict__ D, int M, int K, int Ncols) {
  int colT = Ncols >> 4;
  int colG = colT / CTW;                    // column groups per row
  int kT   = K >> 5;
  int rowT = M >> 4;
  int wave = blockIdx.x * (blockDim.x >> 5) + (threadIdx.x >> 5);
  if (wave >= rowT * colG) return;          // wave-uniform: EXEC all-1s inside
  int rt = wave / colG;
  int ct0 = (wave % colG) * CTW;
  int lane = threadIdx.x & 31;
  int r  = lane & 15;
  int hs = lane >> 4;

  v8f acc[CTW];
#pragma unroll
  for (int j = 0; j < CTW; ++j) acc[j] = (v8f){0.f,0.f,0.f,0.f,0.f,0.f,0.f,0.f};

  const bf16_t* a0 = A + (size_t)(rt * 16 + r) * K + hs * 8;
  const bf16_t* bq = Wpk + ((size_t)ct0 * 32 + lane) * 16;
  size_t bstride = (size_t)colT * 512;      // elements per k-tile

  for (int kt = 0; kt < kT; ++kt) {
    ABu a;
    a.h[0] = *(const v8bf*)(a0);            // K = kt*32 + hs*8 .. +7
    a.h[1] = *(const v8bf*)(a0 + 16);       // K = kt*32 + 16 + hs*8 .. +7
#pragma unroll
    for (int j = 0; j < CTW; ++j) {
      ABu b;
      const bf16_t* bj = bq + (size_t)j * 512;
      b.h[0] = *(const v8bf*)(bj);
      b.h[1] = *(const v8bf*)(bj + 8);
      acc[j] = __builtin_amdgcn_wmma_f32_16x16x32_bf16(
          false, a.v, false, b.v, (short)0, acc[j], false, false);
    }
    a0 += 32;
    bq += bstride;
  }
  // D layout: VGPR j -> row 8*hs + j, col = lane&15
#pragma unroll
  for (int j = 0; j < CTW; ++j) {
    float* drow = D + (size_t)(rt * 16 + hs * 8) * Ncols + ((ct0 + j) << 4) + (lane & 15);
#pragma unroll
    for (int q = 0; q < 8; ++q) drow[(size_t)q * Ncols] = acc[j][q];
  }
}

// ---------------- attention coefficients ----------------
__global__ void k_alpha(const float* __restrict__ Hf,
                        const float* __restrict__ a_src, const float* __restrict__ a_dst,
                        float* as, float* ad, int Nn, int H, int C) {
  int idx = blockIdx.x * blockDim.x + threadIdx.x;
  if (idx >= Nn * H) return;
  int h = idx % H, n = idx / H;
  const float* hv = Hf + (size_t)n * H * C + (size_t)h * C;
  const float* s  = a_src + (size_t)h * C;
  const float* d  = a_dst + (size_t)h * C;
  float accs = 0.f, accd = 0.f;
  for (int c = 0; c < C; ++c) { float x = hv[c]; accs += x * s[c]; accd += x * d[c]; }
  as[idx] = accs;
  ad[idx] = accd;
}

// ---------------- edge kernels ----------------
__device__ __forceinline__ void atomicMaxF(float* addr, float val) {
  if (val >= 0.f) atomicMax((int*)addr, __float_as_int(val));
  else            atomicMin((unsigned int*)addr, __float_as_uint(val));
}

__global__ void k_edge_max(const int* __restrict__ ei, int H,
                           const float* __restrict__ as, const float* __restrict__ ad,
                           float* ev, float* m) {
  long long idx = (long long)blockIdx.x * blockDim.x + threadIdx.x;
  long long tot = (long long)(NEDGES + NNODES) * H;
  if (idx >= tot) return;
  int h = (int)(idx % H);
  long long e = idx / H;
  int s, d;
  if (e < NEDGES) { s = ei[e]; d = ei[NEDGES + e]; }
  else            { s = d = (int)(e - NEDGES); }
  float v = as[(size_t)s * H + h] + ad[(size_t)d * H + h];
  v = (v > 0.f) ? v : NEG_SLOPE * v;
  ev[idx] = v;
  atomicMaxF(&m[(size_t)d * H + h], v);
}

__global__ void k_edge_expsum(const int* __restrict__ ei, int H,
                              const float* __restrict__ m, float* ev, float* denom) {
  long long idx = (long long)blockIdx.x * blockDim.x + threadIdx.x;
  long long tot = (long long)(NEDGES + NNODES) * H;
  if (idx >= tot) return;
  int h = (int)(idx % H);
  long long e = idx / H;
  int d;
  if (e < NEDGES) d = ei[NEDGES + e];
  else            d = (int)(e - NEDGES);
  float v = expf(ev[idx] - m[(size_t)d * H + h]);
  ev[idx] = v;
  atomicAdd(&denom[(size_t)d * H + h], v);
}

__global__ void k_edge_scatter(const int* __restrict__ ei, int H, int C,
                               const float* __restrict__ ev, const float* __restrict__ denom,
                               const float* __restrict__ Hf, float* agg) {
  long long idx = (long long)blockIdx.x * blockDim.x + threadIdx.x;
  long long tot = (long long)(NEDGES + NNODES) * H * C;
  if (idx >= tot) return;
  int c = (int)(idx % C);
  long long eh = idx / C;
  int h = (int)(eh % H);
  long long e = eh / H;
  int s, d;
  if (e < NEDGES) { s = ei[e]; d = ei[NEDGES + e]; }
  else            { s = d = (int)(e - NEDGES); }
  float alpha = ev[eh] / (denom[(size_t)d * H + h] + 1e-16f);
  float msg = Hf[(size_t)s * H * C + (size_t)h * C + c] * alpha;
  atomicAdd(&agg[(size_t)d * H * C + (size_t)h * C + c], msg);
}

// ---------------- epilogues ----------------
__global__ void k_bias_relu(float* agg, const float* __restrict__ b, int HC, size_t n) {
  size_t idx = (size_t)blockIdx.x * blockDim.x + threadIdx.x;
  if (idx >= n) return;
  float v = agg[idx] + b[idx % HC];
  agg[idx] = (v > 0.f) ? v : 0.f;
}

// one wave per node; lane == output channel (OUT_C == 32 == wave32)
__global__ void k_logsoftmax(const float* __restrict__ agg, const float* __restrict__ b,
                             float* out, int Nn) {
  int wave = blockIdx.x * (blockDim.x >> 5) + (threadIdx.x >> 5);
  int lane = threadIdx.x & 31;
  if (wave >= Nn) return;
  float v = agg[(size_t)wave * 32 + lane] + b[lane];
  float m = v;
#pragma unroll
  for (int o = 16; o > 0; o >>= 1) m = fmaxf(m, __shfl_xor(m, o, 32));
  float ex = expf(v - m);
  float s = ex;
#pragma unroll
  for (int o = 16; o > 0; o >>= 1) s += __shfl_xor(s, o, 32);
  out[(size_t)wave * 32 + lane] = v - m - logf(s);
}

// ---------------- host ----------------

static inline unsigned gdiv(size_t n, unsigned b) { return (unsigned)((n + b - 1) / b); }

extern "C" void kernel_launch(void* const* d_in, const int* in_sizes, int n_in,
                              void* d_out, int out_size, void* d_ws, size_t ws_size,
                              hipStream_t stream) {
  (void)in_sizes; (void)n_in; (void)out_size; (void)ws_size;

  const float* x   = (const float*)d_in[0];
  const int*   ei  = (const int*)d_in[1];
  const float* W1  = (const float*)d_in[2];
  const float* a1s = (const float*)d_in[3];
  const float* a1d = (const float*)d_in[4];
  const float* b1  = (const float*)d_in[5];
  const float* W2  = (const float*)d_in[6];
  const float* a2s = (const float*)d_in[7];
  const float* a2d = (const float*)d_in[8];
  const float* b2  = (const float*)d_in[9];
  const float* W3  = (const float*)d_in[10];
  const float* a3s = (const float*)d_in[11];
  const float* a3d = (const float*)d_in[12];
  const float* b3  = (const float*)d_in[13];
  float* out = (float*)d_out;

  // workspace carve-out
  size_t off = 0;
  char* ws = (char*)d_ws;
  auto take = [&](size_t bytes) -> void* {
    void* p = ws + off;
    off += (bytes + 255) & ~(size_t)255;
    return p;
  };
  float*  hbuf0 = (float*) take((size_t)NNODES * HC12 * 4);           // GEMM output h
  float*  hbuf1 = (float*) take((size_t)NNODES * HC12 * 4);           // aggregate
  bf16_t* abf   = (bf16_t*)take((size_t)NNODES * HC12 * 2);           // bf16 activations
  bf16_t* wpk   = (bf16_t*)take((size_t)14 * 28 * 32 * 16 * 2);       // packed W (max)
  float*  asb   = (float*) take((size_t)NNODES * 7 * 4);
  float*  adb   = (float*) take((size_t)NNODES * 7 * 4);
  float*  mb    = (float*) take((size_t)NNODES * 7 * 4);
  float*  db    = (float*) take((size_t)NNODES * 7 * 4);
  float*  evb   = (float*) take((size_t)(NEDGES + NNODES) * 7 * 4);

  auto run_layer = [&](const float* act_in, int K, const float* W,
                       const float* a_s, const float* a_d, const float* b,
                       int H, int C, bool relu) {
    const int HC = H * C;
    // 1. activations -> bf16
    size_t nk = (size_t)NNODES * K;
    k_f32_to_bf16<<<gdiv(nk, 256), 256, 0, stream>>>(act_in, abf, nk);
    // 2. pack W into WMMA B layout
    int kT = K >> 5, nT = HC >> 4;
    k_pack_w<<<gdiv((size_t)kT * nT * 32, 256), 256, 0, stream>>>(W, wpk, K, HC);
    // 3. GEMM h = act @ W  (16x64 strip per wave for wide layers, 16x32 for final)
    if (nT % 4 == 0) {
      int waves = (NNODES >> 4) * (nT / 4);
      k_gemm_wmma<4><<<gdiv((size_t)waves, 8), 256, 0, stream>>>(abf, wpk, hbuf0,
                                                                 NNODES, K, HC);
    } else {
      int waves = (NNODES >> 4) * (nT / 2);
      k_gemm_wmma<2><<<gdiv((size_t)waves, 8), 256, 0, stream>>>(abf, wpk, hbuf0,
                                                                 NNODES, K, HC);
    }
    // 4. attention coefficients
    k_alpha<<<gdiv((size_t)NNODES * H, 256), 256, 0, stream>>>(hbuf0, a_s, a_d, asb, adb,
                                                               NNODES, H, C);
    // 5. init segment buffers
    k_fill<<<gdiv((size_t)NNODES * H, 256), 256, 0, stream>>>(mb, -INFINITY, (size_t)NNODES * H);
    k_fill<<<gdiv((size_t)NNODES * H, 256), 256, 0, stream>>>(db, 0.f, (size_t)NNODES * H);
    k_fill<<<gdiv((size_t)NNODES * HC, 256), 256, 0, stream>>>(hbuf1, 0.f, (size_t)NNODES * HC);
    // 6. edge phase: leaky-relu + segment max, exp + segment sum, weighted scatter
    size_t ne = (size_t)(NEDGES + NNODES) * H;
    k_edge_max   <<<gdiv(ne, 256), 256, 0, stream>>>(ei, H, asb, adb, evb, mb);
    k_edge_expsum<<<gdiv(ne, 256), 256, 0, stream>>>(ei, H, mb, evb, db);
    size_t nec = ne * C;
    k_edge_scatter<<<gdiv(nec, 256), 256, 0, stream>>>(ei, H, C, evb, db, hbuf0, hbuf1);
    // 7. epilogue
    if (relu) {
      k_bias_relu<<<gdiv((size_t)NNODES * HC, 256), 256, 0, stream>>>(hbuf1, b, HC,
                                                                      (size_t)NNODES * HC);
    } else {
      k_logsoftmax<<<gdiv((size_t)NNODES, 8), 256, 0, stream>>>(hbuf1, b, out, NNODES);
    }
  };

  run_layer(x,     IN_C, W1, a1s, a1d, b1, 7, 64, true);   // layer 1
  run_layer(hbuf1, HC12, W2, a2s, a2d, b2, 7, 64, true);   // layer 2
  run_layer(hbuf1, HC12, W3, a3s, a3d, b3, 1, 32, false);  // layer 3 + log_softmax
}
